// RGN_82119774699729
// MI455X (gfx1250) — matrix operand order repaired and checked by reading, not compiled, not run on hardware
//
#include <hip/hip_runtime.h>

// ---------------------------------------------------------------------------
// MI455X (gfx1250, wave32) implementation.
//
// Roofline: only the 5x5 corner of the 32x32 feature map is ever sampled by
// the ROI stage (coords scaled by 0.125 from a [0,31] box => samples in
// [0,3.875]). So the 201MB feats GEMM collapses to 25 rows/batch = 4.9MB of
// HBM traffic. Everything else is KB-scale. We therefore spend WMMA only on
// the real GEMM (f16 inputs / f32 accum, V_WMMA_F32_16X16X32_F16) and keep
// the tiny MLP head in VALU with LDS-resident weights.
// ---------------------------------------------------------------------------

typedef _Float16 v16h __attribute__((ext_vector_type(16)));
typedef float    v8f  __attribute__((ext_vector_type(8)));

#define BSZ    64
#define NP     100
#define NSZ    4
#define CEMB   12
#define KDIM   768
#define ROWS25 25   // 5x5 used region, row25 = h*5+w  <->  n = h*32+w

// ---------------------------------------------------------------------------
// Kernel A: region[b][e][h*5+w] = sum_k feats[b,1+h*32+w,k] * conv_w[e,k] + b[e]
// One wave per 16-row M-tile (2 tiles/batch, 128 waves). K in steps of 32 via
// v_wmma_f32_16x16x32_f16. conv_w staged in LDS as f16 (rows 12..15 zero pad).
//
// A-fragment lane layout (16-bit A 16x32, K order as in B): lane l holds row M=l%16,
//   element j -> K = k0 + (l/16)*16 + j            (one run of 16 contiguous K)
// B-fragment (32x16): lane l holds col N=l%16,
//   element j -> K = k0 + (l/16)*16 + j            (one run of 16 contiguous K)
// C/D f32: lane l element g -> M=(l/16)*8+g, N=l%16.
// ---------------------------------------------------------------------------
__global__ __launch_bounds__(256) void emb_gemm_wmma(
    const float* __restrict__ feats, const float* __restrict__ conv_w,
    const float* __restrict__ conv_b, float* __restrict__ region)
{
  __shared__ __align__(64) _Float16 w16[16 * KDIM];   // 24 KB
  for (int idx = threadIdx.x; idx < 16 * KDIM; idx += 256) {
    int e = idx / KDIM, k = idx % KDIM;
    w16[idx] = (e < CEMB) ? (_Float16)conv_w[e * KDIM + k] : (_Float16)0.0f;
  }
  __syncthreads();

  const int wave  = blockIdx.x * 8 + (threadIdx.x >> 5);   // 0..127
  const int lane  = threadIdx.x & 31;
  const int batch = wave >> 1;
  const int tile  = wave & 1;

  // Row this lane feeds into the A fragment (same row in both lane halves).
  const int mrow  = lane & 15;
  const int row25 = tile * 16 + mrow;
  const int n     = (row25 < ROWS25) ? ((row25 / 5) * 32 + (row25 % 5)) : 0;
  const float* rowp = feats + ((size_t)batch * 1025 + 1 + (size_t)n) * KDIM;

  const int khalf = lane >> 4;   // 0: K-run 0..15; 1: K-run 16..31
  v8f acc = {};

  for (int kk = 0; kk < KDIM / 32; ++kk) {
    const int k0 = kk * 32;
    const int b0 = k0 + khalf * 16;
    float4 f0 = *(const float4*)(rowp + b0);
    float4 f1 = *(const float4*)(rowp + b0 + 4);
    float4 f2 = *(const float4*)(rowp + b0 + 8);
    float4 f3 = *(const float4*)(rowp + b0 + 12);
    v16h a;
    a[0]  = (_Float16)f0.x; a[1]  = (_Float16)f0.y;
    a[2]  = (_Float16)f0.z; a[3]  = (_Float16)f0.w;
    a[4]  = (_Float16)f1.x; a[5]  = (_Float16)f1.y;
    a[6]  = (_Float16)f1.z; a[7]  = (_Float16)f1.w;
    a[8]  = (_Float16)f2.x; a[9]  = (_Float16)f2.y;
    a[10] = (_Float16)f2.z; a[11] = (_Float16)f2.w;
    a[12] = (_Float16)f3.x; a[13] = (_Float16)f3.y;
    a[14] = (_Float16)f3.z; a[15] = (_Float16)f3.w;

    const v16h bfrag =
        *(const v16h*)&w16[(lane & 15) * KDIM + k0 + khalf * 16];

    acc = __builtin_amdgcn_wmma_f32_16x16x32_f16(
        false, a, false, bfrag, (short)0, acc, false, false);
  }

  const int N = lane & 15;
  if (N < CEMB) {
    const float bias  = conv_b[N];
    const int   mbase = khalf * 8;
    for (int g = 0; g < 8; ++g) {
      const int r25 = tile * 16 + mbase + g;
      if (r25 < ROWS25)
        region[((size_t)batch * CEMB + N) * ROWS25 + r25] = acc[g] + bias;
    }
  }
}

// ---------------------------------------------------------------------------
// Kernel B: per batch, mean attention over 12 heads for the 144 valid cells
// (r,c in [10,21]); descending sort (tie-break: higher index first, matching
// argsort(-th) with th strictly increasing in a-rank); first 100 -> points.
// ---------------------------------------------------------------------------
__global__ __launch_bounds__(256) void select_points(
    const float* __restrict__ attn, int* __restrict__ pR, int* __restrict__ pC)
{
  __shared__ float av[144];
  const int b = blockIdx.x;
  const int t = threadIdx.x;
  if (t < 144) {
    const int r = 10 + t / 12, c = 10 + t % 12;
    const int n = r * 32 + c;
    float s = 0.f;
    for (int h = 0; h < 12; ++h)
      s += attn[((size_t)b * 12 + h) * 1025 + 1 + n];
    av[t] = s * (1.0f / 12.0f);
  }
  __syncthreads();
  if (t < 144) {
    const float mine = av[t];
    int rank = 0;
    for (int j = 0; j < 144; ++j) {
      const float o = av[j];
      if (o > mine || (o == mine && j > t)) ++rank;
    }
    if (rank < NP) {
      pR[b * NP + rank] = 10 + t / 12;
      pC[b * NP + rank] = 10 + t % 12;
    }
  }
}

// ---------------------------------------------------------------------------
// Kernel C: ROI-align + MLP head + gumbel argmax + bbox.
// MLP row = b*100 + s*25 + p/4 ; channel = (p%4)*12 + c  (reference reshape).
// Block = 4 rows x 64 spatial threads. All weights + 5x5x12 region in LDS.
// ---------------------------------------------------------------------------
__global__ __launch_bounds__(256) void roi_mlp(
    const float* __restrict__ region,
    const int* __restrict__ pR, const int* __restrict__ pC,
    const float* __restrict__ w1, const float* __restrict__ b1,
    const float* __restrict__ w2, const float* __restrict__ b2,
    const float* __restrict__ w3, const float* __restrict__ b3,
    const float* __restrict__ w4, const float* __restrict__ b4,
    const float* __restrict__ gumbel, float* __restrict__ out)
{
  __shared__ float regL[CEMB * ROWS25];
  __shared__ float w1L[8 * 48], w2L[32], w3L[4 * 256];
  __shared__ float b1L[8], b2L[4], b3L[4], w4L[16], b4L[4];
  __shared__ float red[256 * 4];

  const int b    = blockIdx.x / 25;
  const int rblk = blockIdx.x % 25;
  const int t    = threadIdx.x;

  for (int i = t; i < CEMB * ROWS25; i += 256)
    regL[i] = region[(size_t)b * CEMB * ROWS25 + i];
  for (int i = t; i < 384; i += 256)  w1L[i] = w1[i];
  for (int i = t; i < 1024; i += 256) w3L[i] = w3[i];
  if (t < 32) w2L[t] = w2[t];
  if (t < 16) w4L[t] = w4[t];
  if (t < 8)  b1L[t] = b1[t];
  if (t < 4) { b2L[t] = b2[t]; b3L[t] = b3[t]; b4L[t] = b4[t]; }
  __syncthreads();

  const int q    = t >> 6;          // MLP row within block (0..3)
  const int sp   = t & 63;          // spatial position 0..63
  const int gi   = sp >> 3;         // y-grid index
  const int gj   = sp & 7;          // x-grid index
  const int rib  = rblk * 4 + q;    // row-in-batch 0..99  = s*25 + p/4
  const int s    = rib / 25;
  const int pblk = rib % 25;
  const int rb   = b * 100 + rib;   // global MLP row 0..6399

  const float szf = (float)(4 + 2 * s);

  float h1[8];
  for (int o = 0; o < 8; ++o) h1[o] = b1L[o];

  for (int pq = 0; pq < 4; ++pq) {
    const int p = pblk * 4 + pq;
    const float pr = (float)pR[b * NP + p];
    const float pc = (float)pC[b * NP + p];
    // box = (x1,y1,x2,y2) = clip(pr-sz), clip(pc-sz), clip(pr+sz), clip(pc+sz)
    const float x1 = fminf(fmaxf(pr - szf, 0.f), 31.f);
    const float y1 = fminf(fmaxf(pc - szf, 0.f), 31.f);
    const float x2 = fminf(fmaxf(pr + szf, 0.f), 31.f);
    const float y2 = fminf(fmaxf(pc + szf, 0.f), 31.f);
    const float rw = fmaxf((x2 - x1) * 0.125f, 1.f);
    const float rh = fmaxf((y2 - y1) * 0.125f, 1.f);
    const float ys = y1 * 0.125f + ((float)gi + 0.5f) * rh * 0.125f;
    const float xs = x1 * 0.125f + ((float)gj + 0.5f) * rw * 0.125f;

    const bool inb = (ys >= -1.f) && (ys <= 32.f) && (xs >= -1.f) && (xs <= 32.f);
    const float y = fmaxf(ys, 0.f), x = fmaxf(xs, 0.f);
    const float fy = floorf(y), fx = floorf(x);
    const bool hiy = (fy >= 31.f), hix = (fx >= 31.f);
    int yl = hiy ? 31 : (int)fy;
    int xl = hix ? 31 : (int)fx;
    int yh = min(yl + 1, 31), xh = min(xl + 1, 31);
    const float ly = hiy ? 0.f : (y - fy);
    const float lx = hix ? 0.f : (x - fx);
    const float hy = 1.f - ly, hxw = 1.f - lx;
    // Proven: samples land in [0,3.875] => indices <= 4; clamp defensively.
    yl = min(yl, 4); yh = min(yh, 4); xl = min(xl, 4); xh = min(xh, 4);
    const float w00 = hy * hxw, w01 = hy * lx, w10 = ly * hxw, w11 = ly * lx;

    for (int c = 0; c < CEMB; ++c) {
      const float* rc = &regL[c * ROWS25];
      float v = w00 * rc[yl * 5 + xl] + w01 * rc[yl * 5 + xh] +
                w10 * rc[yh * 5 + xl] + w11 * rc[yh * 5 + xh];
      v = inb ? v : 0.f;
      const int ch = pq * CEMB + c;
      for (int o = 0; o < 8; ++o) h1[o] += v * w1L[o * 48 + ch];
    }
  }

  // conv2: relu(h1) -> h2[4]
  float h2v[4];
  for (int o = 0; o < 4; ++o) {
    float a2 = b2L[o];
    for (int c = 0; c < 8; ++c) a2 += fmaxf(h1[c], 0.f) * w2L[o * 8 + c];
    h2v[o] = fmaxf(a2, 0.f);
  }

  // FC 256->4: feature index f = o*64 + sp; partial sums then LDS tree-reduce.
  for (int k = 0; k < 4; ++k) {
    float a3 = 0.f;
    for (int o = 0; o < 4; ++o) a3 += h2v[o] * w3L[k * 256 + o * 64 + sp];
    red[t * 4 + k] = a3;
  }
  __syncthreads();
  for (int off = 32; off > 0; off >>= 1) {
    if ((t & 63) < off)
      for (int k = 0; k < 4; ++k) red[t * 4 + k] += red[(t + off) * 4 + k];
    __syncthreads();
  }

  if ((t & 63) == 0) {
    float h3[4];
    for (int k = 0; k < 4; ++k) h3[k] = fmaxf(red[t * 4 + k] + b3L[k], 0.f);
    // argmax(log_softmax(h4)+gumbel) == argmax(h4+gumbel); first-max wins ties.
    float best = -3.4e38f; int bi = 0;
    for (int n2 = 0; n2 < 4; ++n2) {
      float v = b4L[n2];
      for (int k = 0; k < 4; ++k) v += h3[k] * w4L[n2 * 4 + k];
      v += gumbel[(size_t)rb * 4 + n2];
      if (v > best) { best = v; bi = n2; }
    }
    const float gap = (float)(4 + 2 * bi);   // SIZES[bi]

    // bbox row rb pairs gap[rb] with pts[rb] = points[rb/100][rb%100].
    const int p2 = rb % 100;
    const float pr2 = (float)pR[b * NP + p2];
    const float pc2 = (float)pC[b * NP + p2];
    out[(size_t)rb * 4 + 0] = fminf(fmaxf(pr2 - gap, 0.f), 31.f);
    out[(size_t)rb * 4 + 1] = fminf(fmaxf(pc2 - gap, 0.f), 31.f);
    out[(size_t)rb * 4 + 2] = fminf(fmaxf(pr2 + gap, 0.f), 31.f);
    out[(size_t)rb * 4 + 3] = fminf(fmaxf(pc2 + gap, 0.f), 31.f);
  }
}

// ---------------------------------------------------------------------------
extern "C" void kernel_launch(void* const* d_in, const int* in_sizes, int n_in,
                              void* d_out, int out_size, void* d_ws, size_t ws_size,
                              hipStream_t stream) {
  const float* feats  = (const float*)d_in[0];
  const float* attn   = (const float*)d_in[1];
  const float* conv_w = (const float*)d_in[2];
  const float* conv_b = (const float*)d_in[3];
  const float* w1 = (const float*)d_in[4];
  const float* b1 = (const float*)d_in[5];
  const float* w2 = (const float*)d_in[6];
  const float* b2 = (const float*)d_in[7];
  const float* w3 = (const float*)d_in[8];
  const float* b3 = (const float*)d_in[9];
  const float* w4 = (const float*)d_in[10];
  const float* b4 = (const float*)d_in[11];
  const float* gumbel = (const float*)d_in[12];
  float* out = (float*)d_out;

  char* ws = (char*)d_ws;
  float* region = (float*)ws;                                 // 64*12*25 f32
  int*   pR     = (int*)(ws + (size_t)BSZ * CEMB * ROWS25 * 4);
  int*   pC     = (int*)(ws + (size_t)BSZ * CEMB * ROWS25 * 4 +
                              (size_t)BSZ * NP * 4);

  emb_gemm_wmma<<<16, 256, 0, stream>>>(feats, conv_w, conv_b, region);
  select_points<<<BSZ, 256, 0, stream>>>(attn, pR, pC);
  roi_mlp<<<BSZ * 25, 256, 0, stream>>>(region, pR, pC,
                                        w1, b1, w2, b2, w3, b3, w4, b4,
                                        gumbel, out);
}
